// GCN_3161095930177
// MI455X (gfx1250) — matrix operand order repaired and checked by reading, not compiled, not run on hardware
//
#include <hip/hip_runtime.h>
#include <hip/hip_bf16.h>
#include <hip/hip_fp16.h>

// ---- problem constants (match reference) ----
#define HDIM   128
#define NHEADS 8
#define DHEAD  16
#define FFDIM  2048
#define BDIM   64      // L (attention length axis)
#define SDIM   1024    // Nb (attention batch axis)
#define NTOK   (BDIM * SDIM)   // 65536 rows
#define NLAY   3
#define FROWS  32      // rows per fused-FFN block

typedef _Float16 v16h __attribute__((ext_vector_type(16)));
typedef _Float16 v8h  __attribute__((ext_vector_type(8)));
typedef float    v8f  __attribute__((ext_vector_type(8)));

// ---------------------------------------------------------------------------
// WMMA fragment helpers (contiguous-k order used by this code for both A and B, wave32)
// A: 16x32 f16 tile, row-major source with leading dim lda.
//   lanes 0-15: row=lane,    K 0..15 contiguous
//   lanes16-31: row=lane-16, K 16..31 contiguous
template <typename PT>
__device__ __forceinline__ v16h load_a_frag(PT A, int lda) {
  const int lane = threadIdx.x & 31;
  const int row  = lane & 15;
  const int koff = (lane & 16) ? 16 : 0;
  PT p = A + (size_t)row * lda + koff;
  v16h r;
#pragma unroll
  for (int i = 0; i < 8; ++i) { r[i] = p[i]; r[i + 8] = p[8 + i]; }
  return r;
}

// B: 32x16 f16 tile where B[k][n] = W[n][k], W row-major [N,K] (weight as stored).
//   lanes 0-15: col=lane,    K 0..15 contiguous;  lanes 16-31: col=lane-16, K 16..31
__device__ __forceinline__ v16h load_b_frag(const _Float16* __restrict__ W, int ldw) {
  const int lane = threadIdx.x & 31;
  const int col  = lane & 15;
  const int kb   = (lane & 16) ? 16 : 0;
  return *reinterpret_cast<const v16h*>(W + (size_t)col * ldw + kb);
}

__device__ __forceinline__ v8f wmma_f16(v16h a, v16h b, v8f c) {
  return __builtin_amdgcn_wmma_f32_16x16x32_f16(false, a, false, b, (short)0, c, false, false);
}

// ---------------------------------------------------------------------------
// C[M,N] = A[M,K] @ W[N,K]^T (+bias)(+relu). Wave -> 16x32 strip, 8 waves/block.
__global__ void k_gemm_f16(const _Float16* __restrict__ A,
                           const _Float16* __restrict__ W,
                           const float* __restrict__ bias,
                           float* __restrict__ C,
                           _Float16* __restrict__ Ch,
                           int M, int N, int K, int relu)
{
  const int wave = threadIdx.x >> 5;
  const int lane = threadIdx.x & 31;
  const int n0 = blockIdx.x * 32;
  const int m0 = (blockIdx.y * 8 + wave) * 16;
  if (m0 >= M) return;                       // wave-uniform: EXEC stays all-ones

  v8f c0 = {};
  v8f c1 = {};
  const _Float16* Abase = A + (size_t)m0 * K;
  const _Float16* W0 = W + (size_t)n0 * K;
  const _Float16* W1 = W + (size_t)(n0 + 16) * K;

  for (int kb = 0; kb < K; kb += 32) {
    if (kb + 64 <= K) __builtin_prefetch(Abase + kb + 32, 0, 3);  // global_prefetch_b8 near
    v16h a  = load_a_frag(Abase + kb, K);
    v16h b0 = load_b_frag(W0 + kb, K);
    v16h b1 = load_b_frag(W1 + kb, K);
    c0 = wmma_f16(a, b0, c0);
    c1 = wmma_f16(a, b1, c1);
  }

  const int nn = lane & 15;
  const int mo = (lane & 16) ? 8 : 0;
  const float bv0 = bias ? bias[n0 + nn]      : 0.0f;
  const float bv1 = bias ? bias[n0 + 16 + nn] : 0.0f;
#pragma unroll
  for (int i = 0; i < 8; ++i) {
    const int m = m0 + mo + i;
    float x0 = c0[i] + bv0;
    float x1 = c1[i] + bv1;
    if (relu) { x0 = fmaxf(x0, 0.0f); x1 = fmaxf(x1, 0.0f); }
    const size_t o0 = (size_t)m * N + n0 + nn;
    const size_t o1 = o0 + 16;
    if (C)  { C[o0] = x0;  C[o1] = x1; }
    if (Ch) { Ch[o0] = (_Float16)x0; Ch[o1] = (_Float16)x1; }
  }
}

// ---------------------------------------------------------------------------
// Fused FFN: af[32rows,128] = relu(h[32,128] @ W1^T + b1) @ W2^T + b2.
// The 32x2048 f16 intermediate lives entirely in LDS (128 KB of the WGP's 320 KB),
// removing a 512 MB/layer HBM round-trip. h tile staged via async global->LDS.
__global__ __launch_bounds__(256) void k_ffn_fused(
    const _Float16* __restrict__ hh,
    const _Float16* __restrict__ w1, const float* __restrict__ b1,
    const _Float16* __restrict__ w2, const float* __restrict__ b2,
    float* __restrict__ af)
{
  __shared__ _Float16 hs[FROWS * HDIM];          // 8 KB staged A tile
  extern __shared__ _Float16 f1s[];              // FROWS*FFDIM f16 = 128 KB

  const int row0 = blockIdx.x * FROWS;
  const int wave = threadIdx.x >> 5;
  const int lane = threadIdx.x & 31;
  const int nn   = lane & 15;
  const int mo   = (lane & 16) ? 8 : 0;

  // ---- async-copy h tile (32x128 f16 = 8 KB) into LDS: 512 x 16B chunks ----
#pragma unroll
  for (int c = 0; c < 2; ++c) {
    const int chunk = (int)threadIdx.x + c * 256;            // 0..511
    const _Float16* g = hh + (size_t)row0 * HDIM + chunk * 8;
    unsigned lds_off = (unsigned)(size_t)(&hs[chunk * 8]);   // low 32 bits = LDS addr
    unsigned long long ga = (unsigned long long)(size_t)g;
    asm volatile("global_load_async_to_lds_b128 %0, %1, off"
                 :: "v"(lds_off), "v"(ga) : "memory");
  }
  asm volatile("s_wait_asynccnt 0x0" ::: "memory");
  __syncthreads();

  // ---- phase 1: f1s[32,2048] = relu(hs @ W1^T + b1) ----
  // wave w: row-tile rt = w&1, 512-wide column slice (w>>1)*512
  {
    const int rt = wave & 1;
    const int colbase = (wave >> 1) * 512;
    v16h a[4];
#pragma unroll
    for (int k = 0; k < 4; ++k)                 // A fragments held in registers
      a[k] = load_a_frag(&hs[rt * 16 * HDIM + k * 32], HDIM);

#pragma unroll 2
    for (int j = 0; j < 16; ++j) {              // 16 col-pairs of 32
      const int n0 = colbase + j * 32;
      const _Float16* W0 = w1 + (size_t)n0 * HDIM;
      const _Float16* W1p = w1 + (size_t)(n0 + 16) * HDIM;
      v8f c0 = {};
      v8f c1 = {};
#pragma unroll
      for (int k = 0; k < 4; ++k) {
        v16h bb0 = load_b_frag(W0 + k * 32, HDIM);
        v16h bb1 = load_b_frag(W1p + k * 32, HDIM);
        c0 = wmma_f16(a[k], bb0, c0);
        c1 = wmma_f16(a[k], bb1, c1);
      }
      const float bv0 = b1[n0 + nn];
      const float bv1 = b1[n0 + 16 + nn];
#pragma unroll
      for (int i = 0; i < 8; ++i) {
        const int m = rt * 16 + mo + i;
        f1s[(size_t)m * FFDIM + n0 + nn]      = (_Float16)fmaxf(c0[i] + bv0, 0.0f);
        f1s[(size_t)m * FFDIM + n0 + 16 + nn] = (_Float16)fmaxf(c1[i] + bv1, 0.0f);
      }
    }
  }
  __syncthreads();

  // ---- phase 2: af[32,128] = f1s @ W2^T + b2 (K = 2048, A from LDS) ----
  // wave w: row-tile rt = w>>2, col-pair (w&3)*32
  {
    const int rt = wave >> 2;
    const int n0 = (wave & 3) * 32;
    const _Float16* W0 = w2 + (size_t)n0 * FFDIM;
    const _Float16* W1p = w2 + (size_t)(n0 + 16) * FFDIM;
    v8f c0 = {};
    v8f c1 = {};
    const _Float16* Albase = &f1s[(size_t)(rt * 16) * FFDIM];
#pragma unroll 4
    for (int kb = 0; kb < FFDIM; kb += 32) {
      v16h a  = load_a_frag(Albase + kb, FFDIM);   // ds_load path
      v16h bb0 = load_b_frag(W0 + kb, FFDIM);
      v16h bb1 = load_b_frag(W1p + kb, FFDIM);
      c0 = wmma_f16(a, bb0, c0);
      c1 = wmma_f16(a, bb1, c1);
    }
    const float bv0 = b2[n0 + nn];
    const float bv1 = b2[n0 + 16 + nn];
#pragma unroll
    for (int i = 0; i < 8; ++i) {
      const int m = row0 + rt * 16 + mo + i;
      af[(size_t)m * HDIM + n0 + nn]      = c0[i] + bv0;
      af[(size_t)m * HDIM + n0 + 16 + nn] = c1[i] + bv1;
    }
  }
}

// ---------------------------------------------------------------------------
// attention: one block per (n, head), 64 threads; thread l owns seq position l.
__global__ void k_attn(const _Float16* __restrict__ qkv, _Float16* __restrict__ ao)
{
  __shared__ float Ks[64 * DHEAD];
  __shared__ float Vs[64 * DHEAD];
  const int n    = blockIdx.x >> 3;
  const int head = blockIdx.x & 7;
  const int l    = threadIdx.x;                   // 0..63

  const size_t rowbase = ((size_t)l * SDIM + n) * (3 * HDIM);
  const int qo = head * DHEAD;
#pragma unroll
  for (int d = 0; d < DHEAD; ++d) {
    Ks[l * DHEAD + d] = (float)qkv[rowbase + HDIM + qo + d];
    Vs[l * DHEAD + d] = (float)qkv[rowbase + 2 * HDIM + qo + d];
  }
  float q[DHEAD];
#pragma unroll
  for (int d = 0; d < DHEAD; ++d) q[d] = (float)qkv[rowbase + qo + d];
  __syncthreads();

  float s[64];
  float mx = -1e30f;
#pragma unroll
  for (int m = 0; m < 64; ++m) {
    float acc = 0.0f;
#pragma unroll
    for (int d = 0; d < DHEAD; ++d) acc += q[d] * Ks[m * DHEAD + d];
    acc *= 0.25f;                                 // 1/sqrt(16)
    s[m] = acc;
    mx = fmaxf(mx, acc);
  }
  float o[DHEAD];
#pragma unroll
  for (int d = 0; d < DHEAD; ++d) o[d] = 0.0f;
  float sum = 0.0f;
#pragma unroll
  for (int m = 0; m < 64; ++m) {
    const float p = __expf(s[m] - mx);
    sum += p;
#pragma unroll
    for (int d = 0; d < DHEAD; ++d) o[d] += p * Vs[m * DHEAD + d];
  }
  const float inv = 1.0f / sum;
  const size_t orow = ((size_t)l * SDIM + n) * HDIM + qo;
#pragma unroll
  for (int d = 0; d < DHEAD; ++d) ao[orow + d] = (_Float16)(o[d] * inv);
}

// ---------------------------------------------------------------------------
__device__ __forceinline__ float wave_sum(float v) {
#pragma unroll
  for (int m = 16; m >= 1; m >>= 1) v += __shfl_xor(v, m, 32);
  return v;
}

// h = LN(h + a)*s + b (in place), also emit f16 copy. One wave per row.
__global__ void k_ln_res(float* __restrict__ h, const float* __restrict__ a,
                         const float* __restrict__ s, const float* __restrict__ b,
                         _Float16* __restrict__ hh)
{
  const int row  = blockIdx.x * 8 + (threadIdx.x >> 5);
  const int lane = threadIdx.x & 31;
  const size_t base = (size_t)row * HDIM;
  float v[4];
#pragma unroll
  for (int j = 0; j < 4; ++j) {
    const int c = lane + 32 * j;
    v[j] = h[base + c] + a[base + c];
  }
  const float mean = wave_sum(v[0] + v[1] + v[2] + v[3]) * (1.0f / HDIM);
  float vs = 0.0f;
#pragma unroll
  for (int j = 0; j < 4; ++j) { const float d = v[j] - mean; vs += d * d; }
  const float var = wave_sum(vs) * (1.0f / HDIM);
  const float rs = rsqrtf(var + 1e-5f);
#pragma unroll
  for (int j = 0; j < 4; ++j) {
    const int c = lane + 32 * j;
    const float y = (v[j] - mean) * rs * s[c] + b[c];
    h[base + c]  = y;
    hh[base + c] = (_Float16)y;
  }
}

// ---------------------------------------------------------------------------
__global__ void k_f32_to_f16(const float* __restrict__ src, _Float16* __restrict__ dst, int n) {
  const int t = blockIdx.x * blockDim.x + threadIdx.x;
  if (t < n) dst[t] = (_Float16)src[t];
}

__global__ void k_add_pos(const float* __restrict__ x, const float* __restrict__ pos,
                          float* __restrict__ h, _Float16* __restrict__ hh)
{
  const size_t t = (size_t)blockIdx.x * blockDim.x + threadIdx.x;  // NTOK*HDIM
  const int r = (int)(t >> 7);
  const int c = (int)(t & 127);
  const int sidx = r & (SDIM - 1);
  const float v = x[t] + pos[(size_t)sidx * HDIM + c];
  h[t] = v;
  hh[t] = (_Float16)v;
}

// ---- GCN helpers ----
__global__ void k_deg_init(float* __restrict__ deg) {
  const int i = blockIdx.x * blockDim.x + threadIdx.x;
  deg[i] = 1.0f;                                  // self-loop contribution
}
__global__ void k_deg_scatter(const int* __restrict__ ei, int E, float* __restrict__ deg) {
  const int t = blockIdx.x * blockDim.x + threadIdx.x;
  if (t < E) atomicAdd(&deg[ei[E + t]], 1.0f);
}
__global__ void k_dis(const float* __restrict__ deg, float* __restrict__ dis) {
  const int i = blockIdx.x * blockDim.x + threadIdx.x;
  dis[i] = rsqrtf(deg[i]);
}
__global__ void k_acc_init(float* __restrict__ acc, const float* __restrict__ bias) {
  const size_t t = (size_t)blockIdx.x * blockDim.x + threadIdx.x;
  acc[t] = bias[t & (HDIM - 1)];
}
// one wave per edge; lane handles 4 strided features (coalesced)
__global__ void k_gcn_scatter(const int* __restrict__ ei, int E,
                              const _Float16* __restrict__ hw,
                              const float* __restrict__ dis,
                              float* __restrict__ acc)
{
  const int e = blockIdx.x * 8 + (threadIdx.x >> 5);
  if (e >= E) return;
  const int lane = threadIdx.x & 31;
  const int src = ei[e];
  const int dst = ei[E + e];
  const float nrm = dis[src] * dis[dst];
  const size_t sb = (size_t)src * HDIM;
  const size_t db = (size_t)dst * HDIM;
#pragma unroll
  for (int j = 0; j < 4; ++j) {
    const int c = lane + 32 * j;
    atomicAdd(&acc[db + c], (float)hw[sb + c] * nrm);
  }
}
__global__ void k_gcn_finalize(const float* __restrict__ acc, const _Float16* __restrict__ hw,
                               const float* __restrict__ dis, _Float16* __restrict__ hh, int relu)
{
  const size_t t = (size_t)blockIdx.x * blockDim.x + threadIdx.x;
  const int i = (int)(t >> 7);
  const float d = dis[i];
  float v = acc[t] + (float)hw[t] * d * d;        // self-loop edge
  if (relu) v = fmaxf(v, 0.0f);
  hh[t] = (_Float16)v;
}
// final 128->1 layer: dot, init, scalar scatter, self-loop
__global__ void k_dot_w3(const _Float16* __restrict__ hh, const float* __restrict__ w3,
                         float* __restrict__ hw3)
{
  const int row  = blockIdx.x * 8 + (threadIdx.x >> 5);
  const int lane = threadIdx.x & 31;
  const size_t base = (size_t)row * HDIM;
  float acc = 0.0f;
#pragma unroll
  for (int j = 0; j < 4; ++j) {
    const int c = lane + 32 * j;
    acc += (float)hh[base + c] * w3[c];
  }
  acc = wave_sum(acc);
  if (lane == 0) hw3[row] = acc;
}
__global__ void k_out_init(float* __restrict__ out, const float* __restrict__ b3) {
  const int i = blockIdx.x * blockDim.x + threadIdx.x;
  out[i] = b3[0];
}
__global__ void k_scatter3(const int* __restrict__ ei, int E, const float* __restrict__ hw3,
                           const float* __restrict__ dis, float* __restrict__ out)
{
  const int t = blockIdx.x * blockDim.x + threadIdx.x;
  if (t < E) {
    const int src = ei[t];
    const int dst = ei[E + t];
    atomicAdd(&out[dst], hw3[src] * dis[src] * dis[dst]);
  }
}
__global__ void k_self3(float* __restrict__ out, const float* __restrict__ hw3,
                        const float* __restrict__ dis)
{
  const int i = blockIdx.x * blockDim.x + threadIdx.x;
  out[i] += hw3[i] * dis[i] * dis[i];
}

// ---------------------------------------------------------------------------
extern "C" void kernel_launch(void* const* d_in, const int* in_sizes, int n_in,
                              void* d_out, int out_size, void* d_ws, size_t ws_size,
                              hipStream_t stream) {
  const float* x    = (const float*)d_in[0];
  const int*   ei   = (const int*)d_in[1];
  const float* pos  = (const float*)d_in[2];
  const float* wqkv = (const float*)d_in[3];
  const float* bqkv = (const float*)d_in[4];
  const float* wo   = (const float*)d_in[5];
  const float* bo   = (const float*)d_in[6];
  const float* ln1s = (const float*)d_in[7];
  const float* ln1b = (const float*)d_in[8];
  const float* w1   = (const float*)d_in[9];
  const float* b1   = (const float*)d_in[10];
  const float* w2   = (const float*)d_in[11];
  const float* b2   = (const float*)d_in[12];
  const float* ln2s = (const float*)d_in[13];
  const float* ln2b = (const float*)d_in[14];
  const float* g1w  = (const float*)d_in[15];
  const float* g1b  = (const float*)d_in[16];
  const float* g2w  = (const float*)d_in[17];
  const float* g2b  = (const float*)d_in[18];
  const float* g3w  = (const float*)d_in[19];
  const float* g3b  = (const float*)d_in[20];
  float* out = (float*)d_out;
  const int E = in_sizes[1] / 2;

  // --- workspace carve ---
  char* p = (char*)d_ws;
  auto alloc = [&](size_t bytes) -> void* {
    void* r = (void*)p;
    p += (bytes + 255) & ~(size_t)255;
    return r;
  };
  float*     h     = (float*)    alloc((size_t)NTOK * HDIM * 4);
  _Float16*  hh    = (_Float16*) alloc((size_t)NTOK * HDIM * 2);
  _Float16*  qkvh  = (_Float16*) alloc((size_t)NTOK * 3 * HDIM * 2);
  _Float16*  aoh   = (_Float16*) alloc((size_t)NTOK * HDIM * 2);
  float*     af    = (float*)    alloc((size_t)NTOK * HDIM * 4);
  _Float16*  hwh   = (_Float16*) alloc((size_t)NTOK * HDIM * 2);
  float*     acc   = (float*)    alloc((size_t)NTOK * HDIM * 4);
  float*     deg   = (float*)    alloc((size_t)NTOK * 4);
  float*     dis   = (float*)    alloc((size_t)NTOK * 4);
  float*     hw3   = (float*)    alloc((size_t)NTOK * 4);
  _Float16*  wqkvh = (_Float16*) alloc((size_t)NLAY * 3 * HDIM * HDIM * 2);
  _Float16*  woh   = (_Float16*) alloc((size_t)NLAY * HDIM * HDIM * 2);
  _Float16*  w1h   = (_Float16*) alloc((size_t)NLAY * FFDIM * HDIM * 2);
  _Float16*  w2h   = (_Float16*) alloc((size_t)NLAY * HDIM * FFDIM * 2);
  _Float16*  g1h   = (_Float16*) alloc((size_t)HDIM * HDIM * 2);
  _Float16*  g2h   = (_Float16*) alloc((size_t)HDIM * HDIM * 2);

  auto cvt = [&](const float* s, _Float16* d, int n) {
    k_f32_to_f16<<<(n + 255) / 256, 256, 0, stream>>>(s, d, n);
  };
  cvt(wqkv, wqkvh, NLAY * 3 * HDIM * HDIM);
  cvt(wo,   woh,   NLAY * HDIM * HDIM);
  cvt(w1,   w1h,   NLAY * FFDIM * HDIM);
  cvt(w2,   w2h,   NLAY * HDIM * FFDIM);
  cvt(g1w,  g1h,   HDIM * HDIM);
  cvt(g2w,  g2h,   HDIM * HDIM);

  auto gemm = [&](const _Float16* A, const _Float16* Wm, const float* bias,
                  float* C, _Float16* Ch, int M, int Nn, int K, int relu) {
    dim3 g(Nn / 32, M / 128);
    k_gemm_f16<<<g, 256, 0, stream>>>(A, Wm, bias, C, Ch, M, Nn, K, relu);
  };

  // h = x + pos
  k_add_pos<<<(NTOK * HDIM) / 256, 256, 0, stream>>>(x, pos, h, hh);

  const size_t ffn_lds = (size_t)FROWS * FFDIM * sizeof(_Float16);  // 128 KB

  for (int i = 0; i < NLAY; ++i) {
    gemm(hh, wqkvh + (size_t)i * 3 * HDIM * HDIM, bqkv + (size_t)i * 3 * HDIM,
         nullptr, qkvh, NTOK, 3 * HDIM, HDIM, 0);
    k_attn<<<SDIM * NHEADS, 64, 0, stream>>>(qkvh, aoh);
    gemm(aoh, woh + (size_t)i * HDIM * HDIM, bo + (size_t)i * HDIM,
         af, nullptr, NTOK, HDIM, HDIM, 0);
    k_ln_res<<<NTOK / 8, 256, 0, stream>>>(h, af, ln1s + (size_t)i * HDIM,
                                           ln1b + (size_t)i * HDIM, hh);
    k_ffn_fused<<<NTOK / FROWS, 256, ffn_lds, stream>>>(
        hh, w1h + (size_t)i * FFDIM * HDIM, b1 + (size_t)i * FFDIM,
        w2h + (size_t)i * HDIM * FFDIM, b2 + (size_t)i * HDIM, af);
    k_ln_res<<<NTOK / 8, 256, 0, stream>>>(h, af, ln2s + (size_t)i * HDIM,
                                           ln2b + (size_t)i * HDIM, hh);
  }

  // ---- GCN ----
  k_deg_init<<<NTOK / 256, 256, 0, stream>>>(deg);
  k_deg_scatter<<<(E + 255) / 256, 256, 0, stream>>>(ei, E, deg);
  k_dis<<<NTOK / 256, 256, 0, stream>>>(deg, dis);

  // layer 1
  gemm(hh, g1h, nullptr, nullptr, hwh, NTOK, HDIM, HDIM, 0);
  k_acc_init<<<(NTOK * HDIM) / 256, 256, 0, stream>>>(acc, g1b);
  k_gcn_scatter<<<(E + 7) / 8, 256, 0, stream>>>(ei, E, hwh, dis, acc);
  k_gcn_finalize<<<(NTOK * HDIM) / 256, 256, 0, stream>>>(acc, hwh, dis, hh, 1);
  // layer 2
  gemm(hh, g2h, nullptr, nullptr, hwh, NTOK, HDIM, HDIM, 0);
  k_acc_init<<<(NTOK * HDIM) / 256, 256, 0, stream>>>(acc, g2b);
  k_gcn_scatter<<<(E + 7) / 8, 256, 0, stream>>>(ei, E, hwh, dis, acc);
  k_gcn_finalize<<<(NTOK * HDIM) / 256, 256, 0, stream>>>(acc, hwh, dis, hh, 1);
  // layer 3 (128 -> 1)
  k_dot_w3<<<NTOK / 8, 256, 0, stream>>>(hh, g3w, hw3);
  k_out_init<<<NTOK / 256, 256, 0, stream>>>(out, g3b);
  k_scatter3<<<(E + 255) / 256, 256, 0, stream>>>(ei, E, hw3, dis, out);
  k_self3<<<NTOK / 256, 256, 0, stream>>>(out, hw3, dis);
}
